// TGDNet_21045339750420
// MI455X (gfx1250) — matrix operand order repaired and checked by reading, not compiled, not run on hardware
//
#include <hip/hip_runtime.h>
#include <hip/hip_bf16.h>

// ---------------------------------------------------------------------------
// CDNA5 / gfx1250 WMMA implementation of the TGD attention block.
// One workgroup per (b, j): 16*32 = 512 blocks, 256 threads (8 wave32).
// Big 1x1 convs -> v_wmma_f32_16x16x32_f16 (f16 operands / f32 accum).
// points2 is pre-transposed to [B,J,NS,C] f16 so each (b,j,chunk) stage is a
// contiguous 16KB block, loaded asynchronously with the Tensor Data Mover
// (tensor_load_to_lds, TENSORcnt) double-buffered against the GEMM chain.
// ---------------------------------------------------------------------------

typedef __attribute__((ext_vector_type(16))) _Float16     v16h;
typedef __attribute__((ext_vector_type(8)))  float        v8f;
typedef __attribute__((ext_vector_type(4)))  unsigned int v4u;
typedef __attribute__((ext_vector_type(8)))  unsigned int v8u;

#define CDIM   256   // channels (K of every big GEMM)
#define JDIM   32    // joints
#define NSAMP  128   // nsample
#define NCH    32    // columns per chunk
#define NCHUNK 4     // NSAMP / NCH
#define KT8    8     // CDIM / 32 k-tiles

__device__ __forceinline__ float lrelu(float x) { return x > 0.f ? x : 0.1f * x; }

__device__ __forceinline__ v8f vzero() {
  v8f z;
#pragma unroll
  for (int i = 0; i < 8; ++i) z[i] = 0.f;
  return z;
}

__device__ __forceinline__ v8f wmma16(v16h a, v16h b, v8f c) {
  // (neg_a, A, neg_b, B, c_mod, C, reuse_a, reuse_b)
  return __builtin_amdgcn_wmma_f32_16x16x32_f16(false, a, false, b, (short)0, c,
                                                false, false);
}

// Issue one TDM copy: 16KB contiguous global -> LDS (2048 x 8-byte elements).
// D# group0: count=1 | lds_addr | global_addr | type=2. group1: data_size=8B,
// tensor_dim0=tile_dim0=2048, tile_dim1=1, stride0=2048.  (2-group form:
// VADDR2/VADDR3 = NULL, tensors up to 2D.)  Tracked by TENSORcnt.
__device__ __forceinline__ void tdm_load_16kb(unsigned int lds_byte,
                                              const void* gptr) {
  unsigned long long ga = (unsigned long long)(uintptr_t)gptr;
  v4u g0;
  g0[0] = 1u;                                   // count=1 (valid), user mode
  g0[1] = lds_byte;                             // lds_addr
  g0[2] = (unsigned int)ga;                     // global_addr[31:0]
  g0[3] = ((unsigned int)(ga >> 32) & 0x01FFFFFFu) | (2u << 30);  // addr[56:32] | type=2
  v8u g1;
  g1[0] = (3u << 16);        // workgroup_mask=0, data_size=3 (8 bytes)
  g1[1] = (2048u << 16);     // tensor_dim0[15:0]=2048   (bits 63:48)
  g1[2] = (1u << 16);        // tensor_dim0 hi=0, tensor_dim1=1
  g1[3] = (2048u << 16);     // tensor_dim1 hi=0, tile_dim0=2048
  g1[4] = 1u;                // tile_dim1=1, tile_dim2=0
  g1[5] = 2048u;             // tensor_dim0_stride lo
  g1[6] = 0u;                // stride hi / dim1_stride lo
  g1[7] = 0u;                // dim1_stride hi
  asm volatile("tensor_load_to_lds %0, %1" ::"s"(g0), "s"(g1) : "memory");
}

// GEMM: out rows [mtBase*16 .. (mtBase+MT)*16), 32 columns, K = 256.
// Wp: packed weights, tile (mt,kt) stored as [32 lanes][16 halves].
// act: LDS activation, column-major: act[n*256 + c], f16.
// UNR: k-tile software-pipeline depth (register pressure control).
template <int MT, int UNR>
__device__ __forceinline__ void gemm_lds(const _Float16* __restrict__ Wp,
                                         const _Float16* act, int mtBase,
                                         int lane, v8f (&acc)[MT][2]) {
  const int csub = lane & 15;
  const int khal = (lane >> 4) << 4;  // 0 or 16
#pragma unroll 1
  for (int kt0 = 0; kt0 < KT8; kt0 += UNR) {
#pragma unroll
    for (int ku = 0; ku < UNR; ++ku) {
      const int kt = kt0 + ku;
      v16h b0 = *(const v16h*)(act + (0 * 16 + csub) * CDIM + kt * 32 + khal);
      v16h b1 = *(const v16h*)(act + (1 * 16 + csub) * CDIM + kt * 32 + khal);
#pragma unroll
      for (int i = 0; i < MT; ++i) {
        v16h a =
            *(const v16h*)(Wp + (((mtBase + i) * KT8 + kt) * 32 + lane) * 16);
        acc[i][0] = wmma16(a, b0, acc[i][0]);
        acc[i][1] = wmma16(a, b1, acc[i][1]);
      }
    }
  }
}

// ---------------------------------------------------------------------------
// Weight packing: fp32 [O x K] -> f16 WMMA A-fragment lane order.
// ---------------------------------------------------------------------------
__global__ __launch_bounds__(256) void pack_w(const float* __restrict__ W,
                                              _Float16* __restrict__ dst, int O,
                                              int K) {
  int gid = blockIdx.x * 256 + threadIdx.x;
  int KT = K >> 5;
  int total = (O >> 4) * KT * 32;
  if (gid >= total) return;
  int lane = gid & 31;
  int tidx = gid >> 5;  // mt*KT + kt
  int kt = tidx % KT;
  int mt = tidx / KT;
  int m = mt * 16 + (lane & 15);
  int khalf = (lane >> 4) * 16;
  _Float16* d = dst + ((size_t)tidx * 32 + lane) * 16;
  const float* w = W + (size_t)m * K + kt * 32;
#pragma unroll
  for (int i = 0; i < 8; ++i) d[i] = (_Float16)w[khalf + i];
#pragma unroll
  for (int i = 0; i < 8; ++i) d[8 + i] = (_Float16)w[khalf + 8 + i];
}

// ---------------------------------------------------------------------------
// points2 [B,C,NS,J] f32  ->  X16 [B,J,NS,C] f16 (coalesced via LDS tile).
// One block per (b, ns).
// ---------------------------------------------------------------------------
__global__ __launch_bounds__(256) void xpose(const float* __restrict__ points2,
                                             _Float16* __restrict__ X16) {
  __shared__ _Float16 tile[JDIM * CDIM];  // 16 KB
  const int b = blockIdx.x >> 7;   // / NSAMP
  const int ns = blockIdx.x & 127; // % NSAMP
  const int t = threadIdx.x;
  for (int idx = t; idx < JDIM * CDIM; idx += 256) {
    const int c = idx >> 5;
    const int jj = idx & 31;
    tile[jj * CDIM + c] =
        (_Float16)points2[((size_t)(b * CDIM + c) * NSAMP + ns) * JDIM + jj];
  }
  __syncthreads();
  for (int idx = t; idx < JDIM * CDIM; idx += 256) {
    const int jj = idx >> 8;
    const int c = idx & 255;
    X16[(((size_t)(b * JDIM + jj) * NSAMP) + ns) * CDIM + c] =
        tile[jj * CDIM + c];
  }
}

// ---------------------------------------------------------------------------
// Precompute: pg = einsum('bcn,cnm->bcm'), p1g = lrelu(Wg@pg + bg),
// fq = Wfq@p1g, fk = Wfk@p1g.  One block per batch.
// ---------------------------------------------------------------------------
__global__ __launch_bounds__(256) void precompute(
    const float* __restrict__ points1, const float* __restrict__ graph_a,
    const float* __restrict__ Wg, const float* __restrict__ bg,
    const float* __restrict__ Wfq, const float* __restrict__ Wfk,
    float* __restrict__ p1g_ws, float* __restrict__ fq_ws,
    float* __restrict__ fk_ws) {
  __shared__ float pgs[CDIM * JDIM];   // 32 KB
  __shared__ float p1gs[CDIM * JDIM];  // 32 KB
  const int b = blockIdx.x;
  const int t = threadIdx.x;

  float pr[JDIM];
#pragma unroll
  for (int n = 0; n < JDIM; ++n)
    pr[n] = points1[((size_t)b * CDIM + t) * JDIM + n];
  float pgr[JDIM];
#pragma unroll
  for (int m = 0; m < JDIM; ++m) pgr[m] = 0.f;
  for (int n = 0; n < JDIM; ++n) {
    const float pv = pr[n];
    const float* ga = graph_a + ((size_t)t * JDIM + n) * JDIM;
#pragma unroll
    for (int m = 0; m < JDIM; ++m) pgr[m] += pv * ga[m];
  }
#pragma unroll
  for (int m = 0; m < JDIM; ++m) pgs[t * JDIM + m] = pgr[m];
  __syncthreads();

  for (int j = 0; j < JDIM; ++j) {
    float s = bg[t];
    const float* wg = Wg + (size_t)t * CDIM;
    for (int c = 0; c < CDIM; ++c) s += wg[c] * pgs[c * JDIM + j];
    s = lrelu(s);
    p1gs[t * JDIM + j] = s;
    p1g_ws[((size_t)b * CDIM + t) * JDIM + j] = s;
  }
  __syncthreads();

  for (int j = 0; j < JDIM; ++j) {
    float sq = 0.f, sk = 0.f;
    const float* wq = Wfq + (size_t)t * CDIM;
    const float* wk = Wfk + (size_t)t * CDIM;
    for (int c = 0; c < CDIM; ++c) {
      const float pv = p1gs[c * JDIM + j];
      sq += wq[c] * pv;
      sk += wk[c] * pv;
    }
    fq_ws[((size_t)b * CDIM + t) * JDIM + j] = sq;
    fk_ws[((size_t)b * CDIM + t) * JDIM + j] = sk;
  }
}

// ---------------------------------------------------------------------------
// Main fused kernel: one block per (b, j).
// ---------------------------------------------------------------------------
__global__ __launch_bounds__(256)
__attribute__((amdgpu_waves_per_eu(4))) void tgd_main(
    const float* __restrict__ points1, const float* __restrict__ points2,
    const _Float16* __restrict__ X16, const float* __restrict__ p1g_ws,
    const float* __restrict__ fq_ws, const float* __restrict__ fk_ws,
    const _Float16* __restrict__ Wq0p, const _Float16* __restrict__ Wq1p,
    const _Float16* __restrict__ Wq2p, const _Float16* __restrict__ Wk0p,
    const _Float16* __restrict__ Wk1p, const _Float16* __restrict__ Wfv1p,
    const _Float16* __restrict__ Wfv2p, const _Float16* __restrict__ Wv0p,
    const _Float16* __restrict__ Wv1p, const float* __restrict__ bq0,
    const float* __restrict__ bq1, const float* __restrict__ bq2,
    const float* __restrict__ bk0, const float* __restrict__ bk1,
    const float* __restrict__ bk2, const float* __restrict__ Wk2,
    const float* __restrict__ bv0, const float* __restrict__ bv1,
    const float* __restrict__ Wv2, const float* __restrict__ bv2,
    float* __restrict__ out, int use_x16) {
  __shared__ __align__(32) _Float16 actX[NCH * CDIM];  // 16 KB staged X chunk
  __shared__ __align__(32) _Float16 actA[NCH * CDIM];  // 16 KB ping
  __shared__ __align__(32) _Float16 actB[NCH * CDIM];  // 16 KB pong
  __shared__ float p1gL[CDIM], fqL[CDIM], fkL[CDIM], p1cL[CDIM];
  __shared__ float k2L[2 * CDIM];
  __shared__ float redL[CDIM];  // kmax then vmax
  __shared__ float vresL[CDIM];

  const int blk = blockIdx.x;
  const int b = blk >> 5;
  const int j = blk & 31;
  const int t = threadIdx.x;
  const int wave = t >> 5;
  const int lane = t & 31;
  const int rsub = (lane >> 4) << 3;  // row sub-offset in tile (0 or 8)
  const int csub = lane & 15;         // column within N-tile

  const unsigned int ldsX = (unsigned int)(uintptr_t)(&actX[0]);
  const _Float16* xsrc = X16 + (size_t)(b * JDIM + j) * NSAMP * CDIM;

  // broadcast vectors for this (b, j)
  p1gL[t] = p1g_ws[((size_t)b * CDIM + t) * JDIM + j];
  fqL[t] = fq_ws[((size_t)b * CDIM + t) * JDIM + j];
  fkL[t] = fk_ws[((size_t)b * CDIM + t) * JDIM + j];
  p1cL[t] = points1[((size_t)b * CDIM + t) * JDIM + j];

  // fallback staging: gather straight from points2 (strided)
  auto stage_manual = [&](int cc) {
    for (int idx = t; idx < CDIM * NCH; idx += 256) {
      const int c = idx >> 5;
      const int n = idx & 31;
      actX[n * CDIM + c] = (_Float16)
          points2[((size_t)(b * CDIM + c) * NSAMP + cc * NCH + n) * JDIM + j];
    }
  };
  auto tdm_issue = [&](int cc) {
    if (use_x16 && wave == 0)
      tdm_load_16kb(ldsX, xsrc + (size_t)cc * NCH * CDIM);
  };
  auto stage_wait = [&](int cc) {
    if (use_x16) {
      __builtin_amdgcn_s_wait_tensorcnt(0);
    } else {
      stage_manual(cc);
    }
    __syncthreads();
  };

  tdm_issue(0);  // prefetch first chunk

  // ======================= Phase 1: K branch =======================
  float kmx[2][8];
#pragma unroll
  for (int i = 0; i < 2; ++i)
#pragma unroll
    for (int v = 0; v < 8; ++v) kmx[i][v] = -1e30f;

  for (int cc = 0; cc < NCHUNK; ++cc) {
    stage_wait(cc);

    v8f k0[2][2];
#pragma unroll
    for (int i = 0; i < 2; ++i) { k0[i][0] = vzero(); k0[i][1] = vzero(); }
    gemm_lds<2, 4>(Wk0p, actX, wave * 2, lane, k0);
    __syncthreads();
    // actX free: prefetch next chunk (chunk 0 again when rolling into phase 2)
    tdm_issue(cc < NCHUNK - 1 ? cc + 1 : 0);
#pragma unroll
    for (int i = 0; i < 2; ++i)
#pragma unroll
      for (int nt = 0; nt < 2; ++nt)
#pragma unroll
        for (int v = 0; v < 8; ++v) {
          const int row = (wave * 2 + i) * 16 + rsub + v;
          const int col = nt * 16 + csub;
          const float val = lrelu(k0[i][nt][v] + bk0[row] + fkL[row]);
          actB[col * CDIM + row] = (_Float16)val;
        }
    __syncthreads();

    v8f k1[2][2];
#pragma unroll
    for (int i = 0; i < 2; ++i) { k1[i][0] = vzero(); k1[i][1] = vzero(); }
    gemm_lds<2, 4>(Wk1p, actB, wave * 2, lane, k1);
#pragma unroll
    for (int i = 0; i < 2; ++i)
#pragma unroll
      for (int nt = 0; nt < 2; ++nt)
#pragma unroll
        for (int v = 0; v < 8; ++v) {
          const int row = (wave * 2 + i) * 16 + rsub + v;
          const float val = lrelu(k1[i][nt][v] + bk1[row]);
          kmx[i][v] = fmaxf(kmx[i][v], val);
        }
    __syncthreads();
  }

  // reduce kmax across the 16 column-lanes of each half, publish to LDS
#pragma unroll
  for (int i = 0; i < 2; ++i)
#pragma unroll
    for (int v = 0; v < 8; ++v) {
#pragma unroll
      for (int m = 1; m < 16; m <<= 1)
        kmx[i][v] = fmaxf(kmx[i][v], __shfl_xor(kmx[i][v], m, 32));
      if (csub == 0) redL[(wave * 2 + i) * 16 + rsub + v] = kmx[i][v];
    }
  __syncthreads();

  // k2 = Wk2 @ kmax + bk2  (512x256 matvec, 2 rows per thread)
  for (int r = t; r < 2 * CDIM; r += 256) {
    float s = bk2[r];
    const float* w = Wk2 + (size_t)r * CDIM;
    for (int c = 0; c < CDIM; ++c) s += w[c] * redL[c];
    k2L[r] = s;
  }
  __syncthreads();

  // ======================= Phase 2: Q + V branches =======================
  float vsum[2][8];
  float vmx[2][8];
#pragma unroll
  for (int i = 0; i < 2; ++i)
#pragma unroll
    for (int v = 0; v < 8; ++v) { vsum[i][v] = 0.f; vmx[i][v] = -1e30f; }

  for (int cc = 0; cc < NCHUNK; ++cc) {
    stage_wait(cc);

    // V0 = Wfv1 @ X (keep in registers through the Q chain)
    v8f v0a[2][2];
#pragma unroll
    for (int i = 0; i < 2; ++i) { v0a[i][0] = vzero(); v0a[i][1] = vzero(); }
    gemm_lds<2, 4>(Wfv1p, actX, wave * 2, lane, v0a);

    // Q0
    v8f q0a[2][2];
#pragma unroll
    for (int i = 0; i < 2; ++i) { q0a[i][0] = vzero(); q0a[i][1] = vzero(); }
    gemm_lds<2, 4>(Wq0p, actX, wave * 2, lane, q0a);
    __syncthreads();
    // actX free: prefetch next chunk while the rest of the chain runs
    if (cc < NCHUNK - 1) tdm_issue(cc + 1);
#pragma unroll
    for (int i = 0; i < 2; ++i)
#pragma unroll
      for (int nt = 0; nt < 2; ++nt)
#pragma unroll
        for (int v = 0; v < 8; ++v) {
          const int row = (wave * 2 + i) * 16 + rsub + v;
          const int col = nt * 16 + csub;
          const float val = lrelu(q0a[i][nt][v] + bq0[row] + fqL[row]);
          actB[col * CDIM + row] = (_Float16)val;
        }
    __syncthreads();

    // Q1
    v8f q1a[2][2];
#pragma unroll
    for (int i = 0; i < 2; ++i) { q1a[i][0] = vzero(); q1a[i][1] = vzero(); }
    gemm_lds<2, 4>(Wq1p, actB, wave * 2, lane, q1a);
    __syncthreads();
#pragma unroll
    for (int i = 0; i < 2; ++i)
#pragma unroll
      for (int nt = 0; nt < 2; ++nt)
#pragma unroll
        for (int v = 0; v < 8; ++v) {
          const int row = (wave * 2 + i) * 16 + rsub + v;
          const int col = nt * 16 + csub;
          const float val = lrelu(q1a[i][nt][v] + bq1[row]);
          actA[col * CDIM + row] = (_Float16)val;
        }
    __syncthreads();

    // Q2 (512 rows) + sigmoid gate -> g1 (rows<256) in actB, g2*p1g in actA
    v8f q2a[4][2];
#pragma unroll
    for (int i = 0; i < 4; ++i) { q2a[i][0] = vzero(); q2a[i][1] = vzero(); }
    gemm_lds<4, 2>(Wq2p, actA, wave * 4, lane, q2a);
    __syncthreads();
#pragma unroll
    for (int i = 0; i < 4; ++i)
#pragma unroll
      for (int nt = 0; nt < 2; ++nt)
#pragma unroll
        for (int v = 0; v < 8; ++v) {
          const int row = (wave * 4 + i) * 16 + rsub + v;
          const int col = nt * 16 + csub;
          const float q2 = q2a[i][nt][v] + bq2[row];
          const float aa = 1.f / (1.f + __expf(-k2L[row] * q2));
          if (row < CDIM) {
            actB[col * CDIM + row] = (_Float16)aa;              // g1
          } else {
            const int r2 = row - CDIM;
            actA[col * CDIM + r2] = (_Float16)(aa * p1gL[r2]);  // g2 * p1g
          }
        }
    __syncthreads();

    // pe = Wfv2 @ (g2*p1g)
    v8f pea[2][2];
#pragma unroll
    for (int i = 0; i < 2; ++i) { pea[i][0] = vzero(); pea[i][1] = vzero(); }
    gemm_lds<2, 4>(Wfv2p, actA, wave * 2, lane, pea);
    __syncthreads();

    // v = lrelu(V0*g1 + pe) + points1 ; accumulate mean ; store v -> actA
    float g1c[2][2][8];
#pragma unroll
    for (int i = 0; i < 2; ++i)
#pragma unroll
      for (int nt = 0; nt < 2; ++nt)
#pragma unroll
        for (int v = 0; v < 8; ++v) {
          const int row = (wave * 2 + i) * 16 + rsub + v;
          const int col = nt * 16 + csub;
          const float g1 = (float)actB[col * CDIM + row];
          g1c[i][nt][v] = g1;
          const float vv =
              lrelu(v0a[i][nt][v] * g1 + pea[i][nt][v]) + p1cL[row];
          vsum[i][v] += vv;
          actA[col * CDIM + row] = (_Float16)vv;
        }
    __syncthreads();

    // v2 = lrelu((Wv0@v + bv0)*g1 + pe) -> actA
    v8f w0a[2][2];
#pragma unroll
    for (int i = 0; i < 2; ++i) { w0a[i][0] = vzero(); w0a[i][1] = vzero(); }
    gemm_lds<2, 4>(Wv0p, actA, wave * 2, lane, w0a);
    __syncthreads();
#pragma unroll
    for (int i = 0; i < 2; ++i)
#pragma unroll
      for (int nt = 0; nt < 2; ++nt)
#pragma unroll
        for (int v = 0; v < 8; ++v) {
          const int row = (wave * 2 + i) * 16 + rsub + v;
          const int col = nt * 16 + csub;
          const float val = w0a[i][nt][v] + bv0[row];
          const float v2 = lrelu(val * g1c[i][nt][v] + pea[i][nt][v]);
          actA[col * CDIM + row] = (_Float16)v2;
        }
    __syncthreads();

    // V1 = lrelu(Wv1@v2 + bv1); running max over ns
    v8f v1a[2][2];
#pragma unroll
    for (int i = 0; i < 2; ++i) { v1a[i][0] = vzero(); v1a[i][1] = vzero(); }
    gemm_lds<2, 4>(Wv1p, actA, wave * 2, lane, v1a);
#pragma unroll
    for (int i = 0; i < 2; ++i)
#pragma unroll
      for (int nt = 0; nt < 2; ++nt)
#pragma unroll
        for (int v = 0; v < 8; ++v) {
          const int row = (wave * 2 + i) * 16 + rsub + v;
          const float val = lrelu(v1a[i][nt][v] + bv1[row]);
          vmx[i][v] = fmaxf(vmx[i][v], val);
        }
    __syncthreads();
  }

  // reduce vmax / vsum across column-lanes, publish
#pragma unroll
  for (int i = 0; i < 2; ++i)
#pragma unroll
    for (int v = 0; v < 8; ++v) {
#pragma unroll
      for (int m = 1; m < 16; m <<= 1) {
        vmx[i][v] = fmaxf(vmx[i][v], __shfl_xor(vmx[i][v], m, 32));
        vsum[i][v] += __shfl_xor(vsum[i][v], m, 32);
      }
      if (csub == 0) {
        const int row = (wave * 2 + i) * 16 + rsub + v;
        redL[row] = vmx[i][v];
        vresL[row] = vsum[i][v] * (1.f / (float)NSAMP);
      }
    }
  __syncthreads();

  // out[o] = lrelu(Wv2@vmax + bv2) + v_res   (one row per thread)
  {
    const int o = t;
    float s = bv2[o];
    const float* w = Wv2 + (size_t)o * CDIM;
    for (int c = 0; c < CDIM; ++c) s += w[c] * redL[c];
    s = lrelu(s) + vresL[o];
    out[((size_t)b * CDIM + o) * JDIM + j] = s;
  }
}

// ---------------------------------------------------------------------------
// Host-side launch
// ---------------------------------------------------------------------------
extern "C" void kernel_launch(void* const* d_in, const int* in_sizes, int n_in,
                              void* d_out, int out_size, void* d_ws,
                              size_t ws_size, hipStream_t stream) {
  (void)in_sizes; (void)n_in; (void)out_size;
  const float* points1 = (const float*)d_in[2];
  const float* points2 = (const float*)d_in[3];
  const float* graph_a = (const float*)d_in[4];
  const float* Wg = (const float*)d_in[5];
  const float* bg = (const float*)d_in[6];
  const float* Wfq = (const float*)d_in[7];
  const float* Wfk = (const float*)d_in[8];
  const float* Wfv1 = (const float*)d_in[9];
  const float* Wfv2 = (const float*)d_in[10];
  const float* Wq0 = (const float*)d_in[11];
  const float* bq0 = (const float*)d_in[12];
  const float* Wq1 = (const float*)d_in[13];
  const float* bq1 = (const float*)d_in[14];
  const float* Wq2 = (const float*)d_in[15];
  const float* bq2 = (const float*)d_in[16];
  const float* Wk0 = (const float*)d_in[17];
  const float* bk0 = (const float*)d_in[18];
  const float* Wk1 = (const float*)d_in[19];
  const float* bk1 = (const float*)d_in[20];
  const float* Wk2 = (const float*)d_in[21];
  const float* bk2 = (const float*)d_in[22];
  const float* Wv0 = (const float*)d_in[23];
  const float* bv0 = (const float*)d_in[24];
  const float* Wv1 = (const float*)d_in[25];
  const float* bv1 = (const float*)d_in[26];
  const float* Wv2 = (const float*)d_in[27];
  const float* bv2 = (const float*)d_in[28];
  float* out = (float*)d_out;

  char* ws = (char*)d_ws;
  size_t off = 0;
  auto carve = [&](size_t bytes) -> void* {
    void* p = ws + off;
    off += (bytes + 255) & ~(size_t)255;
    return p;
  };
  const size_t wsz = (size_t)256 * 256 * sizeof(_Float16);
  _Float16* Wq0p = (_Float16*)carve(wsz);
  _Float16* Wq1p = (_Float16*)carve(wsz);
  _Float16* Wq2p = (_Float16*)carve(2 * wsz);
  _Float16* Wk0p = (_Float16*)carve(wsz);
  _Float16* Wk1p = (_Float16*)carve(wsz);
  _Float16* Wfv1p = (_Float16*)carve(wsz);
  _Float16* Wfv2p = (_Float16*)carve(wsz);
  _Float16* Wv0p = (_Float16*)carve(wsz);
  _Float16* Wv1p = (_Float16*)carve(wsz);
  const size_t vsz = (size_t)16 * 256 * 32 * sizeof(float);
  float* p1g_ws = (float*)carve(vsz);
  float* fq_ws = (float*)carve(vsz);
  float* fk_ws = (float*)carve(vsz);
  _Float16* X16 =
      (_Float16*)carve((size_t)16 * 32 * 128 * 256 * sizeof(_Float16));
  const int use_x16 = (off <= ws_size) ? 1 : 0;

  const int thr256 = (256 / 16) * (256 / 32) * 32;  // 4096
  const int thr512 = (512 / 16) * (256 / 32) * 32;  // 8192
  pack_w<<<(thr256 + 255) / 256, 256, 0, stream>>>(Wq0, Wq0p, 256, 256);
  pack_w<<<(thr256 + 255) / 256, 256, 0, stream>>>(Wq1, Wq1p, 256, 256);
  pack_w<<<(thr512 + 255) / 256, 256, 0, stream>>>(Wq2, Wq2p, 512, 256);
  pack_w<<<(thr256 + 255) / 256, 256, 0, stream>>>(Wk0, Wk0p, 256, 256);
  pack_w<<<(thr256 + 255) / 256, 256, 0, stream>>>(Wk1, Wk1p, 256, 256);
  pack_w<<<(thr256 + 255) / 256, 256, 0, stream>>>(Wfv1, Wfv1p, 256, 256);
  pack_w<<<(thr256 + 255) / 256, 256, 0, stream>>>(Wfv2, Wfv2p, 256, 256);
  pack_w<<<(thr256 + 255) / 256, 256, 0, stream>>>(Wv0, Wv0p, 256, 256);
  pack_w<<<(thr256 + 255) / 256, 256, 0, stream>>>(Wv1, Wv1p, 256, 256);

  if (use_x16) {
    xpose<<<16 * 128, 256, 0, stream>>>(points2, X16);
  }
  precompute<<<16, 256, 0, stream>>>(points1, graph_a, Wg, bg, Wfq, Wfk,
                                     p1g_ws, fq_ws, fk_ws);

  tgd_main<<<16 * 32, 256, 0, stream>>>(
      points1, points2, X16, p1g_ws, fq_ws, fk_ws, Wq0p, Wq1p, Wq2p, Wk0p,
      Wk1p, Wfv1p, Wfv2p, Wv0p, Wv1p, bq0, bq1, bq2, bk0, bk1, bk2, Wk2, bv0,
      bv1, Wv2, bv2, out, use_x16);
}
